// EnergyPredictor_31001073943174
// MI455X (gfx1250) — matrix operand order repaired and checked by reading, not compiled, not run on hardware
//
#include <hip/hip_runtime.h>
#include <math.h>

// ---------------------------------------------------------------------------
// Types for CDNA5 WMMA (wave32): 16x16x32 bf16 -> f32
// ---------------------------------------------------------------------------
typedef __attribute__((ext_vector_type(16))) __bf16         v16bf;
typedef __attribute__((ext_vector_type(16))) unsigned short v16u;
typedef __attribute__((ext_vector_type(4)))  unsigned short v4u;
typedef __attribute__((ext_vector_type(8)))  float          v8f;

#define FOLD     256
#define HALF     128
#define NGR      64
#define WAVES    8      // waves per block in the scores kernel (256 threads)
#define POOLROWS 2048   // rows per block in the pool kernel

__device__ __forceinline__ unsigned short f2bf(float f) {
  unsigned int u = __float_as_uint(f);
  u += 0x7FFFu + ((u >> 16) & 1u);     // round-to-nearest-even
  return (unsigned short)(u >> 16);
}

__device__ __forceinline__ void atomicMaxF(float* addr, float val) {
  if (val >= 0.0f) atomicMax((int*)addr, __float_as_int(val));
  else             atomicMin((unsigned int*)addr, __float_as_uint(val));
}

// ---------------------------------------------------------------------------
// Convert W1 fp32 [256][128] -> bf16 pre-swizzled into WMMA B-fragment order:
// out[((kk*8 + nt)*32 + lane)*16 + i] = bf16(W1[kk*32 + (lane>>4)*8 + (i&7) + (i>>3)*16]
//                                              [nt*16 + (lane&15)])
// so each lane's 16 B-fragment elements are 32 contiguous bytes.
// ---------------------------------------------------------------------------
__global__ void cvt_swz_kernel(const float* __restrict__ W1,
                               unsigned short* __restrict__ out) {
  int o = blockIdx.x * blockDim.x + threadIdx.x;
  if (o < 8 * 8 * 32 * 16) {
    int i  = o & 15;
    int l  = (o >> 4) & 31;
    int nt = (o >> 9) & 7;
    int kk = o >> 12;
    int col = l & 15, hi = l >> 4;
    int k = kk * 32 + hi * 8 + (i & 7) + ((i >> 3) << 4);
    out[o] = f2bf(W1[(size_t)k * HALF + nt * 16 + col]);
  }
}

// ---------------------------------------------------------------------------
// Init segment-max / denom / combined-pool accumulators
// ---------------------------------------------------------------------------
__global__ void init_kernel(float* __restrict__ gmax, float* __restrict__ denom,
                            float* __restrict__ comb) {
  int i = blockIdx.x * blockDim.x + threadIdx.x;
  if (i < 3 * NGR) { gmax[i] = -INFINITY; denom[i] = 0.0f; }
  if (i < NGR * 3 * FOLD) comb[i] = 0.0f;
}

// ---------------------------------------------------------------------------
// Attention scores: s = relu(x @ W1 + b1) @ W2 + b2  via bf16 WMMA.
// Each wave owns a 16-row tile. x tile is converted+swizzled into LDS in
// A-fragment order (wide ds stores/loads); W1 B-fragments come pre-swizzled
// from global (WGP$-hot, 64 KB). Per K-step: all 8 B fragments are batched
// into registers first, then 8 back-to-back v_wmma_f32_16x16x32_bf16 retire
// against staggered loadcnt waits. b1/ReLU/W2 folded on the f32 accumulators
// + 16-lane shfl reduction. No __syncthreads: each wave only touches its own
// LDS slice.
// ---------------------------------------------------------------------------
__global__ __launch_bounds__(32 * WAVES)
void scores_kernel(const float* __restrict__ x,
                   const unsigned short* __restrict__ w1swz,  // [8][8][32][16] bf16
                   const float* __restrict__ b1,              // [128]
                   const float* __restrict__ w2,              // [128]
                   const float* __restrict__ b2s,             // [1]
                   float* __restrict__ scores, int N) {
  // A tiles in fragment order: [wave][kchunk][lane][16 bf16]  (64 KB total)
  __shared__ __align__(32) unsigned short aSwz[WAVES][8][32][16];

  const int t  = threadIdx.x;
  const int wv = t >> 5;
  const int l  = t & 31;
  const int hi  = l >> 4;      // lane half (ISA wave32 fragment layout)
  const int col = l & 15;
  const int tileBase = (blockIdx.x * WAVES + wv) * 16;

  // Stage + convert + swizzle this wave's 16x256 A tile into LDS.
  // Lane loads float4 (coalesced b128), stores 4 bf16 as one b64.
  // A-fragment k-map: element i -> klocal = (i<8 ? i : i+8) + hi*8; a 4-aligned
  // group of klocal never crosses a segment, so 4 elems stay contiguous.
  for (int i4 = l; i4 < 16 * 64; i4 += 32) {
    int r  = i4 >> 6;
    int c0 = (i4 & 63) << 2;
    int row = tileBase + r;
    float4 v = make_float4(0.0f, 0.0f, 0.0f, 0.0f);
    if (row < N) v = *(const float4*)(x + (size_t)row * FOLD + c0);
    int kk = c0 >> 5, klocal = c0 & 31;
    int h  = (klocal >> 3) & 1;                     // which lane-half owns it
    int i0 = (klocal & 7) + ((klocal >> 4) << 3);   // fragment element base
    v4u pk;
    pk[0] = f2bf(v.x); pk[1] = f2bf(v.y); pk[2] = f2bf(v.z); pk[3] = f2bf(v.w);
    *(v4u*)&aSwz[wv][kk][h * 16 + r][i0] = pk;
  }

  v8f acc[8];
  const v8f zero = {0.0f, 0.0f, 0.0f, 0.0f, 0.0f, 0.0f, 0.0f, 0.0f};
#pragma unroll
  for (int nt = 0; nt < 8; ++nt) acc[nt] = zero;

  for (int kk = 0; kk < 8; ++kk) {           // K = 256 in chunks of 32
    v16u au = *(const v16u*)&aSwz[wv][kk][l][0];          // 2x ds_load_b128
    v16bf a = __builtin_bit_cast(v16bf, au);
    // Batch all 8 B fragments (16x global_load_b128 in one clause) ...
    v16u bu[8];
#pragma unroll
    for (int nt = 0; nt < 8; ++nt)
      bu[nt] = *(const v16u*)(w1swz + ((size_t)((kk * 8 + nt) * 32 + l) << 4));
    // ... then fire 8 WMMAs back-to-back with staggered loadcnt waits.
#pragma unroll
    for (int nt = 0; nt < 8; ++nt) {
      v16bf b = __builtin_bit_cast(v16bf, bu[nt]);
      acc[nt] = __builtin_amdgcn_wmma_f32_16x16x32_bf16(
          false, a, false, b, (short)0, acc[nt], false, false);
    }
  }

  // s[M] = sum_N relu(hidden[M][N] + b1[N]) * w2[N] + b2
  float part[8];
#pragma unroll
  for (int r = 0; r < 8; ++r) part[r] = 0.0f;
#pragma unroll
  for (int nt = 0; nt < 8; ++nt) {
    float bb = b1[nt * 16 + col];
    float ww = w2[nt * 16 + col];
#pragma unroll
    for (int r = 0; r < 8; ++r) {
      float h = acc[nt][r] + bb;
      h = h > 0.0f ? h : 0.0f;
      part[r] += h * ww;
    }
  }
  // Reduce across the 16 lanes of each half (C layout: N = lane&15, M = r + 8*hi).
#pragma unroll
  for (int r = 0; r < 8; ++r)
#pragma unroll
    for (int m = 1; m < 16; m <<= 1) part[r] += __shfl_xor(part[r], m, 32);

  if (col == 0) {
    float bias2 = b2s[0];
#pragma unroll
    for (int r = 0; r < 8; ++r) {
      int row = tileBase + r + hi * 8;
      if (row < N) scores[row] = part[r] + bias2;
    }
  }
}

// ---------------------------------------------------------------------------
// Segment softmax pieces (64 graphs live in L2; float atomics).
// segmax also materializes per-row graph ids into gbuf so the (possibly
// gathered) index lookup happens exactly once.
// ---------------------------------------------------------------------------
__global__ void segmax_kernel(const float* __restrict__ s, const int* __restrict__ batch,
                              const int* __restrict__ gather, int* __restrict__ gbuf,
                              float* __restrict__ gmax, int N) {
  int i = blockIdx.x * blockDim.x + threadIdx.x;
  if (i < N) {
    int g = gather ? batch[gather[i]] : batch[i];
    gbuf[i] = g;
    atomicMaxF(&gmax[g], s[i]);
  }
}

__global__ void sumexp_kernel(float* __restrict__ s, const int* __restrict__ gbuf,
                              const float* __restrict__ gmax,
                              float* __restrict__ denom, int N) {
  int i = blockIdx.x * blockDim.x + threadIdx.x;
  if (i < N) {
    int g = gbuf[i];
    float m = gmax[g];
    if (!(m > -INFINITY)) m = 0.0f;             // empty-segment guard
    float e = expf(s[i] - m);
    s[i] = e;                                   // overwrite score with exp value
    atomicAdd(&denom[g], e);
  }
}

// ---------------------------------------------------------------------------
// Weighted segment-sum: each block owns a [64 graphs][256 cols] LDS
// accumulator and a POOLROWS-row chunk. Thread t owns column t: it
// accumulates in a register while the row's graph id is unchanged (batches
// are sorted for protein/ligand), spills to its private LDS cell on a
// change (covers the unsorted cross ids), and flushes once per block with
// skip-zero global atomics.
// ---------------------------------------------------------------------------
__global__ __launch_bounds__(256)
void pool_kernel(const float* __restrict__ x, const float* __restrict__ e,
                 const int* __restrict__ gbuf, const float* __restrict__ denom,
                 float* __restrict__ comb, int colOff, int N) {
  __shared__ float accs[NGR * FOLD];   // 64 KB
  const int t = threadIdx.x;
  for (int i = t; i < NGR * FOLD; i += 256) accs[i] = 0.0f;
  __syncthreads();

  const int base = blockIdx.x * POOLROWS;
  const int end  = (base + POOLROWS < N) ? base + POOLROWS : N;
  float racc = 0.0f;
  int   gcur = -1;
  for (int row = base; row < end; ++row) {
    int   g = gbuf[row];                 // row-uniform -> scalarizable
    float w = e[row] / denom[g];
    float v = x[(size_t)row * FOLD + t] * w;
    if (g == gcur) {
      racc += v;
    } else {
      if (gcur >= 0) accs[gcur * FOLD + t] += racc;
      gcur = g;
      racc = v;
    }
  }
  if (gcur >= 0) accs[gcur * FOLD + t] += racc;
  __syncthreads();

  for (int i = t; i < NGR * FOLD; i += 256) {
    float v = accs[i];
    if (v != 0.0f)
      atomicAdd(&comb[(size_t)(i >> 8) * (3 * FOLD) + colOff + (i & 255)], v);
  }
}

// ---------------------------------------------------------------------------
// Final MLP: [64,768] -> relu 256 -> relu 128 -> 1.  One block, trivial cost.
// ---------------------------------------------------------------------------
__global__ __launch_bounds__(256)
void final_mlp_kernel(const float* __restrict__ comb,
                      const float* __restrict__ W1, const float* __restrict__ b1,
                      const float* __restrict__ W2, const float* __restrict__ b2,
                      const float* __restrict__ oW, const float* __restrict__ ob,
                      float* __restrict__ out) {
  __shared__ float h1[FOLD];
  __shared__ float h2[HALF];
  __shared__ float red[8];
  const int t = threadIdx.x;
  for (int g = 0; g < NGR; ++g) {
    float a = b1[t];
    for (int k = 0; k < 3 * FOLD; ++k) a += comb[g * 3 * FOLD + k] * W1[(size_t)k * FOLD + t];
    h1[t] = a > 0.0f ? a : 0.0f;
    __syncthreads();
    if (t < HALF) {
      float a2 = b2[t];
      for (int k = 0; k < FOLD; ++k) a2 += h1[k] * W2[(size_t)k * HALF + t];
      h2[t] = a2 > 0.0f ? a2 : 0.0f;
    }
    __syncthreads();
    float v = (t < HALF) ? h2[t] * oW[t] : 0.0f;
    for (int m = 1; m < 32; m <<= 1) v += __shfl_xor(v, m, 32);
    if ((t & 31) == 0) red[t >> 5] = v;
    __syncthreads();
    if (t == 0) {
      float s = ob[0];
      for (int w = 0; w < 8; ++w) s += red[w];
      out[g] = s;
    }
    __syncthreads();
  }
}

// ---------------------------------------------------------------------------
// Launch
// ---------------------------------------------------------------------------
static inline size_t align256(size_t x) { return (x + 255) & ~(size_t)255; }

extern "C" void kernel_launch(void* const* d_in, const int* in_sizes, int n_in,
                              void* d_out, int out_size, void* d_ws, size_t ws_size,
                              hipStream_t stream) {
  const float* rec    = (const float*)d_in[0];
  const float* lig    = (const float*)d_in[1];
  const float* cross  = (const float*)d_in[2];
  const int*   cidx0  = (const int*)d_in[3];     // row 0 of cross_idx
  const int*   pbatch = (const int*)d_in[4];
  const int*   lbatch = (const int*)d_in[5];
  const float* paW1 = (const float*)d_in[7],  *pab1 = (const float*)d_in[8];
  const float* paW2 = (const float*)d_in[9],  *pab2 = (const float*)d_in[10];
  const float* laW1 = (const float*)d_in[11], *lab1 = (const float*)d_in[12];
  const float* laW2 = (const float*)d_in[13], *lab2 = (const float*)d_in[14];
  const float* caW1 = (const float*)d_in[15], *cab1 = (const float*)d_in[16];
  const float* caW2 = (const float*)d_in[17], *cab2 = (const float*)d_in[18];
  const float* mW1  = (const float*)d_in[19], *mb1  = (const float*)d_in[20];
  const float* mW2  = (const float*)d_in[21], *mb2  = (const float*)d_in[22];
  const float* oW   = (const float*)d_in[23], *obi  = (const float*)d_in[24];
  float* out = (float*)d_out;

  const int NP = in_sizes[0] / FOLD;
  const int NL = in_sizes[1] / FOLD;
  const int NC = in_sizes[2] / FOLD;

  // Workspace carve-up
  char* ws = (char*)d_ws;
  size_t off = 0;
  unsigned short* w1swz_p = (unsigned short*)(ws + off); off = align256(off + FOLD * HALF * 2);
  unsigned short* w1swz_l = (unsigned short*)(ws + off); off = align256(off + FOLD * HALF * 2);
  unsigned short* w1swz_c = (unsigned short*)(ws + off); off = align256(off + FOLD * HALF * 2);
  float* sPro  = (float*)(ws + off); off = align256(off + (size_t)NP * 4);
  float* sLig  = (float*)(ws + off); off = align256(off + (size_t)NL * 4);
  float* sCro  = (float*)(ws + off); off = align256(off + (size_t)NC * 4);
  int*   gPro  = (int*)(ws + off);   off = align256(off + (size_t)NP * 4);
  int*   gLig  = (int*)(ws + off);   off = align256(off + (size_t)NL * 4);
  int*   gCro  = (int*)(ws + off);   off = align256(off + (size_t)NC * 4);
  float* gmax  = (float*)(ws + off); off = align256(off + 3 * NGR * 4);
  float* denom = (float*)(ws + off); off = align256(off + 3 * NGR * 4);
  float* comb  = (float*)(ws + off); off = align256(off + (size_t)NGR * 3 * FOLD * 4);
  (void)ws_size; (void)n_in; (void)out_size;

  const int WN = FOLD * HALF;
  cvt_swz_kernel<<<(WN + 255) / 256, 256, 0, stream>>>(paW1, w1swz_p);
  cvt_swz_kernel<<<(WN + 255) / 256, 256, 0, stream>>>(laW1, w1swz_l);
  cvt_swz_kernel<<<(WN + 255) / 256, 256, 0, stream>>>(caW1, w1swz_c);

  init_kernel<<<(NGR * 3 * FOLD + 255) / 256, 256, 0, stream>>>(gmax, denom, comb);

  // Scores via WMMA (one 16-row tile per wave, WAVES waves per block)
  auto sgrid = [](int N) { return ((N + 15) / 16 + WAVES - 1) / WAVES; };
  scores_kernel<<<sgrid(NP), 32 * WAVES, 0, stream>>>(rec,   w1swz_p, pab1, paW2, pab2, sPro, NP);
  scores_kernel<<<sgrid(NL), 32 * WAVES, 0, stream>>>(lig,   w1swz_l, lab1, laW2, lab2, sLig, NL);
  scores_kernel<<<sgrid(NC), 32 * WAVES, 0, stream>>>(cross, w1swz_c, cab1, caW2, cab2, sCro, NC);

  // Segment max (+ materialize per-row graph ids)
  segmax_kernel<<<(NP + 255) / 256, 256, 0, stream>>>(sPro, pbatch, nullptr, gPro, gmax + 0 * NGR, NP);
  segmax_kernel<<<(NL + 255) / 256, 256, 0, stream>>>(sLig, lbatch, nullptr, gLig, gmax + 1 * NGR, NL);
  segmax_kernel<<<(NC + 255) / 256, 256, 0, stream>>>(sCro, lbatch, cidx0,   gCro, gmax + 2 * NGR, NC);

  // exp + denominators (scores buffers become exp values)
  sumexp_kernel<<<(NP + 255) / 256, 256, 0, stream>>>(sPro, gPro, gmax + 0 * NGR, denom + 0 * NGR, NP);
  sumexp_kernel<<<(NL + 255) / 256, 256, 0, stream>>>(sLig, gLig, gmax + 1 * NGR, denom + 1 * NGR, NL);
  sumexp_kernel<<<(NC + 255) / 256, 256, 0, stream>>>(sCro, gCro, gmax + 2 * NGR, denom + 2 * NGR, NC);

  // Weighted segment-sum into combined [64, 768] via block-local LDS accum
  auto pgrid = [](int N) { return (N + POOLROWS - 1) / POOLROWS; };
  pool_kernel<<<pgrid(NP), 256, 0, stream>>>(rec,   sPro, gPro, denom + 0 * NGR, comb, 0 * FOLD, NP);
  pool_kernel<<<pgrid(NL), 256, 0, stream>>>(lig,   sLig, gLig, denom + 1 * NGR, comb, 1 * FOLD, NL);
  pool_kernel<<<pgrid(NC), 256, 0, stream>>>(cross, sCro, gCro, denom + 2 * NGR, comb, 2 * FOLD, NC);

  // Final energy MLP
  final_mlp_kernel<<<1, 256, 0, stream>>>(comb, mW1, mb1, mW2, mb2, oW, obi, out);
}
